// enhance_net_55327768708642
// MI455X (gfx1250) — matrix operand order repaired and checked by reading, not compiled, not run on hardware
//
#include <hip/hip_runtime.h>

typedef _Float16 f16;
typedef __attribute__((ext_vector_type(16))) _Float16 v16h;
typedef __attribute__((ext_vector_type(8)))  _Float16 h8;
typedef __attribute__((ext_vector_type(4)))  _Float16 h4;
typedef __attribute__((ext_vector_type(8)))  float    v8f;

static __device__ __forceinline__ float preluf(float v, float a) {
    return v >= 0.f ? v : a * v;
}

// ---------------------------------------------------------------------------
// Utility kernels
// ---------------------------------------------------------------------------
__global__ void k_zero_f32(float* __restrict__ p, long n) {
    for (long i = blockIdx.x * (long)blockDim.x + threadIdx.x; i < n;
         i += (long)gridDim.x * blockDim.x)
        p[i] = 0.f;
}

__global__ void k_zero_f16(f16* __restrict__ p, long n) {
    for (long i = blockIdx.x * (long)blockDim.x + threadIdx.x; i < n;
         i += (long)gridDim.x * blockDim.x)
        p[i] = (f16)0.f;
}

// cast a f32 NCHW tensor into a channel slice of a padded f16 NHWC tensor
__global__ void k_cast_slice_f16(const float* __restrict__ src, f16* __restrict__ dst,
                                 int B, int C, int Cdst, int c0, int HW) {
    long n = (long)B * C * HW;
    for (long i = blockIdx.x * (long)blockDim.x + threadIdx.x; i < n;
         i += (long)gridDim.x * blockDim.x) {
        int p = (int)(i % HW);
        long t = i / HW;
        int c = (int)(t % C);
        int b = (int)(t / C);
        dst[((long)b * HW + p) * Cdst + c0 + c] = (f16)src[i];
    }
}

// ---------------------------------------------------------------------------
// Curve stage: r_k = x^n_k / (x^n_k + sigma_k^n_k + bias_k), per-(k,c) BN
// ---------------------------------------------------------------------------
__global__ void k_curve_stats(const float* __restrict__ x, const float* __restrict__ n16,
                              const float* __restrict__ sig, const float* __restrict__ bia,
                              float* __restrict__ stats, int B, int HW) {
    const int c = blockIdx.y;          // 0..2
    const long N = (long)B * HW;
    float s[16], s2[16];
#pragma unroll
    for (int k = 0; k < 16; ++k) { s[k] = 0.f; s2[k] = 0.f; }
    for (long i = blockIdx.x * (long)blockDim.x + threadIdx.x; i < N;
         i += (long)gridDim.x * blockDim.x) {
        long b = i / HW, p = i % HW;
        float xv = x[(b * 3 + c) * HW + p];
#pragma unroll
        for (int k = 0; k < 16; ++k) {
            float nk = n16[k];
            float xn = powf(xv, nk);
            float r  = xn / (xn + powf(sig[k], nk) + bia[k]);
            s[k]  += r;
            s2[k] += r * r;
        }
    }
    __shared__ float red[256];
    for (int k = 0; k < 16; ++k) {
        red[threadIdx.x] = s[k];
        __syncthreads();
        for (int off = 128; off > 0; off >>= 1) {
            if ((int)threadIdx.x < off) red[threadIdx.x] += red[threadIdx.x + off];
            __syncthreads();
        }
        if (threadIdx.x == 0) atomicAdd(&stats[(k * 3 + c) * 2 + 0], red[0]);
        __syncthreads();
        red[threadIdx.x] = s2[k];
        __syncthreads();
        for (int off = 128; off > 0; off >>= 1) {
            if ((int)threadIdx.x < off) red[threadIdx.x] += red[threadIdx.x + off];
            __syncthreads();
        }
        if (threadIdx.x == 0) atomicAdd(&stats[(k * 3 + c) * 2 + 1], red[0]);
        __syncthreads();
    }
}

// stats[c] = {sum, sumsq}  ->  stats[c] = {scale, shift}
__global__ void k_bn_finalize(float* __restrict__ stats, const float* __restrict__ g,
                              const float* __restrict__ b, int C, int gmod, float invN) {
    int c = threadIdx.x;
    if (c >= C) return;
    float mean = stats[c * 2] * invN;
    float var  = stats[c * 2 + 1] * invN - mean * mean;
    float sc   = g[c % gmod] * rsqrtf(var + 1e-5f);
    stats[c * 2]     = sc;
    stats[c * 2 + 1] = b[c % gmod] - mean * sc;
}

// recompute curves, normalize, write f16 x_fusion NHWC (channel = k*3+c, Cpad=64)
__global__ void k_curve_apply(const float* __restrict__ x, const float* __restrict__ n16,
                              const float* __restrict__ sig, const float* __restrict__ bia,
                              const float* __restrict__ stats, f16* __restrict__ dst,
                              int B, int HW) {
    const int c = blockIdx.y;
    const long N = (long)B * HW;
    for (long i = blockIdx.x * (long)blockDim.x + threadIdx.x; i < N;
         i += (long)gridDim.x * blockDim.x) {
        long b = i / HW, p = i % HW;
        float xv = x[(b * 3 + c) * HW + p];
        f16* drow = dst + ((long)b * HW + p) * 64;
#pragma unroll
        for (int k = 0; k < 16; ++k) {
            float nk = n16[k];
            float xn = powf(xv, nk);
            float r  = xn / (xn + powf(sig[k], nk) + bia[k]);
            int ch = k * 3 + c;
            float y = r * stats[ch * 2] + stats[ch * 2 + 1];
            drow[ch] = (f16)y;
        }
    }
}

// ---------------------------------------------------------------------------
// Weight prep: f32 OIHW (3x3) -> f16 [tap][O][Ipad] (zero-padded input chans)
// ---------------------------------------------------------------------------
__global__ void k_wprep(const float* __restrict__ w, f16* __restrict__ wp,
                        int O, int I, int Ipad) {
    long n = 9L * O * Ipad;
    for (long i = blockIdx.x * (long)blockDim.x + threadIdx.x; i < n;
         i += (long)gridDim.x * blockDim.x) {
        int ip = (int)(i % Ipad);
        long t2 = i / Ipad;
        int o = (int)(t2 % O);
        int t = (int)(t2 / O);
        float v = (ip < I) ? w[(long)(o * I + ip) * 9 + t] : 0.f;
        wp[i] = (f16)v;
    }
}

// ---------------------------------------------------------------------------
// Implicit-GEMM 3x3 conv (pad=1, stride=1) via V_WMMA_F32_16X16X32_F16.
// Input activations: f16 NHWC ([B][H][W][Cpad]).  Weights: f16 [tap][Cout][Cpad].
// Output: f32 NCHW.  Block = 256 threads = 8 waves; block tile = 32 chan x 128
// pixels; wave tile = 16 x 32 (A fragment reused across two WMMAs).
// Out-of-bounds taps redirect the load pointer to a zero page (branchless).
// ---------------------------------------------------------------------------
__global__ void __launch_bounds__(256)
k_conv3x3_wmma(const f16* __restrict__ in, const f16* __restrict__ wts,
               const float* __restrict__ bias, const f16* __restrict__ zbuf,
               float* __restrict__ out,
               int B, int Cpad, int Cout, int H, int W) {
    __shared__ f16 lA[32 * 32];    // weights tile: 32 Cout x 32 K     (2 KB)
    __shared__ f16 lB[128 * 32];   // im2col tile:  128 pixels x 32 K  (8 KB)

    const int tid    = threadIdx.x;
    const int lane   = tid & 31;
    const int waveId = tid >> 5;
    const int m_off  = (waveId >> 2) * 16;   // 0 / 16
    const int n_base = (waveId & 3) * 32;    // 0 / 32 / 64 / 96

    const int  HWl   = H * W;
    const long Npix  = (long)B * HWl;
    const long pix0  = (long)blockIdx.x * 128;
    const int  cout0 = blockIdx.y * 32;

    v8f acc0 = {0.f, 0.f, 0.f, 0.f, 0.f, 0.f, 0.f, 0.f};
    v8f acc1 = {0.f, 0.f, 0.f, 0.f, 0.f, 0.f, 0.f, 0.f};

    // B-staging assignment: this thread stages pixel (tid>>1), channels (tid&1)*16..+15
    const int pl   = tid >> 1;          // 0..127
    const int kh16 = (tid & 1) * 16;
    const long pB  = pix0 + pl;
    int by = 0, yy = 0, xx = 0, pin = 0;
    if (pB < Npix) {
        long b = pB / HWl, r = pB % HWl;
        by = (int)b; yy = (int)(r / W); xx = (int)(r % W); pin = 1;
    }

    const int nCb = Cpad >> 5;
    for (int t = 0; t < 9; ++t) {
        const int ky = t / 3 - 1, kx = t % 3 - 1;
        const int iy = yy + ky, ix = xx + kx;
        const bool ok = pin && (iy >= 0) && (iy < H) && (ix >= 0) && (ix < W);
        const f16* srcB =
            ok ? (in + ((long)by * HWl + (long)iy * W + ix) * Cpad + kh16) : zbuf;
        if (t < 8)  // warm L2 with next tap's weights
            __builtin_prefetch(wts + ((long)(t + 1) * Cout + cout0) * Cpad, 0, 1);
        const f16* wsrc = wts + ((long)t * Cout + cout0) * Cpad;

        for (int cb = 0; cb < nCb; ++cb) {
            __syncthreads();
            // --- stage A tile (32x32), one b64 per thread ---
            {
                const int mi = tid >> 3;
                const int ki = (tid * 4) & 31;
                *(h4*)&lA[tid * 4] = *(const h4*)(wsrc + (long)mi * Cpad + cb * 32 + ki);
            }
            // --- stage B tile (128x32, pixel-major), two b128 per thread ---
            {
                const f16* s = srcB + cb * 32;   // zbuf is large enough for any cb
                *(h8*)&lB[pl * 32 + kh16]     = *(const h8*)(s);
                *(h8*)&lB[pl * 32 + kh16 + 8] = *(const h8*)(s + 8);
            }
            __syncthreads();
            // --- fragments: contiguous 16B runs -> ds_load_b128 ---
            const int khalf = lane >> 4;
            const int ml    = lane & 15;
            union { v16h v; h8 h[2]; } af, bf0, bf1;
            const f16* ar = &lA[(m_off + ml) * 32 + khalf * 8];
            af.h[0] = *(const h8*)(ar);
            af.h[1] = *(const h8*)(ar + 16);
            const f16* br0 = &lB[(n_base + ml) * 32 + khalf * 8];
            bf0.h[0] = *(const h8*)(br0);
            bf0.h[1] = *(const h8*)(br0 + 16);
            const f16* br1 = &lB[(n_base + 16 + ml) * 32 + khalf * 8];
            bf1.h[0] = *(const h8*)(br1);
            bf1.h[1] = *(const h8*)(br1 + 16);
            acc0 = __builtin_amdgcn_wmma_f32_16x16x32_f16(
                false, af.v, false, bf0.v, (short)0, acc0, false, false);
            acc1 = __builtin_amdgcn_wmma_f32_16x16x32_f16(
                false, af.v, false, bf1.v, (short)0, acc1, false, false);
        }
    }

    // --- epilogue: C/D layout lane = N, vgpr j -> M = (lane>=16)*8 + j ---
    const int nl = lane & 15;
    union { v8f v; float e[8]; } a0, a1;
    a0.v = acc0;
    a1.v = acc1;
    long p0 = pix0 + n_base + nl;
    if (p0 < Npix) {
        long b = p0 / HWl, r = p0 % HWl;
#pragma unroll
        for (int j = 0; j < 8; ++j) {
            int oc = cout0 + m_off + (lane >> 4) * 8 + j;
            out[((long)b * Cout + oc) * HWl + r] = a0.e[j] + bias[oc];
        }
    }
    long p1 = pix0 + n_base + 16 + nl;
    if (p1 < Npix) {
        long b = p1 / HWl, r = p1 % HWl;
#pragma unroll
        for (int j = 0; j < 8; ++j) {
            int oc = cout0 + m_off + (lane >> 4) * 8 + j;
            out[((long)b * Cout + oc) * HWl + r] = a1.e[j] + bias[oc];
        }
    }
}

// ---------------------------------------------------------------------------
// PReLU + 2x2 maxpool (stride 2)
// ---------------------------------------------------------------------------
__global__ void k_prelu_pool(const float* __restrict__ in, float* __restrict__ out,
                             const float* __restrict__ pa, int B, int C, int H, int W) {
    const int Ho = H >> 1, Wo = W >> 1;
    const long n = (long)B * C * Ho * Wo;
    const float a = pa[0];
    for (long i = blockIdx.x * (long)blockDim.x + threadIdx.x; i < n;
         i += (long)gridDim.x * blockDim.x) {
        int xo = (int)(i % Wo);
        long t = i / Wo;
        int yo = (int)(t % Ho); t /= Ho;
        int c  = (int)(t % C);
        int b  = (int)(t / C);
        long base = ((long)b * C + c) * H * W + (long)(2 * yo) * W + 2 * xo;
        float v00 = preluf(in[base], a);
        float v01 = preluf(in[base + 1], a);
        float v10 = preluf(in[base + W], a);
        float v11 = preluf(in[base + W + 1], a);
        out[i] = fmaxf(fmaxf(v00, v01), fmaxf(v10, v11));
    }
}

// ---------------------------------------------------------------------------
// Per-channel sum/sumsq over (N,H,W)
// ---------------------------------------------------------------------------
__global__ void k_chan_stats(const float* __restrict__ x, float* __restrict__ stats,
                             int B, int C, int HW) {
    const int c = blockIdx.y;
    const long N = (long)B * HW;
    float s = 0.f, s2 = 0.f;
    for (long i = blockIdx.x * (long)blockDim.x + threadIdx.x; i < N;
         i += (long)gridDim.x * blockDim.x) {
        long b = i / HW, p = i % HW;
        float v = x[(b * C + c) * HW + p];
        s += v; s2 += v * v;
    }
    __shared__ float red[256];
    red[threadIdx.x] = s;
    __syncthreads();
    for (int off = 128; off > 0; off >>= 1) {
        if ((int)threadIdx.x < off) red[threadIdx.x] += red[threadIdx.x + off];
        __syncthreads();
    }
    if (threadIdx.x == 0) atomicAdd(&stats[c * 2], red[0]);
    __syncthreads();
    red[threadIdx.x] = s2;
    __syncthreads();
    for (int off = 128; off > 0; off >>= 1) {
        if ((int)threadIdx.x < off) red[threadIdx.x] += red[threadIdx.x + off];
        __syncthreads();
    }
    if (threadIdx.x == 0) atomicAdd(&stats[c * 2 + 1], red[0]);
}

__global__ void k_bn_apply(const float* __restrict__ in, float* __restrict__ out,
                           const float* __restrict__ stats, long n, int C, int HW) {
    for (long i = blockIdx.x * (long)blockDim.x + threadIdx.x; i < n;
         i += (long)gridDim.x * blockDim.x) {
        int c = (int)((i / HW) % C);
        out[i] = in[i] * stats[c * 2] + stats[c * 2 + 1];
    }
}

// ---------------------------------------------------------------------------
// PReLU -> bilinear 2x upsample (align_corners=True) -> optional sigmoid
// ---------------------------------------------------------------------------
__global__ void k_up2_prelu(const float* __restrict__ in, float* __restrict__ out,
                            const float* __restrict__ pa, int B, int C, int H, int W,
                            int sigm) {
    const int Ho = H * 2, Wo = W * 2;
    const long n = (long)B * C * Ho * Wo;
    const float a = pa[0];
    const float sy = (float)(H - 1) / (float)(Ho - 1);
    const float sx = (float)(W - 1) / (float)(Wo - 1);
    for (long i = blockIdx.x * (long)blockDim.x + threadIdx.x; i < n;
         i += (long)gridDim.x * blockDim.x) {
        int xo = (int)(i % Wo);
        long t = i / Wo;
        int yo = (int)(t % Ho); t /= Ho;
        int c  = (int)(t % C);
        int b  = (int)(t / C);
        float py = yo * sy; int ylo = (int)py; float fy = py - ylo;
        int yhi = min(ylo + 1, H - 1);
        float px = xo * sx; int xlo = (int)px; float fx = px - xlo;
        int xhi = min(xlo + 1, W - 1);
        const float* src = in + ((long)b * C + c) * H * W;
        float v00 = preluf(src[(long)ylo * W + xlo], a);
        float v01 = preluf(src[(long)ylo * W + xhi], a);
        float v10 = preluf(src[(long)yhi * W + xlo], a);
        float v11 = preluf(src[(long)yhi * W + xhi], a);
        float v = (1.f - fy) * ((1.f - fx) * v00 + fx * v01)
                +        fy  * ((1.f - fx) * v10 + fx * v11);
        if (sigm) v = 1.f / (1.f + __expf(-v));
        out[i] = v;
    }
}

// ---------------------------------------------------------------------------
// 1x1 conv 32 -> 3 with bias
// ---------------------------------------------------------------------------
__global__ void k_conv1x1_3(const float* __restrict__ in, const float* __restrict__ w,
                            const float* __restrict__ b, float* __restrict__ out,
                            int B, int Cin, int HW) {
    const long n = (long)B * HW;
    for (long i = blockIdx.x * (long)blockDim.x + threadIdx.x; i < n;
         i += (long)gridDim.x * blockDim.x) {
        long bb = i / HW, p = i % HW;
        const float* src = in + bb * Cin * HW + p;
        float a0 = b[0], a1 = b[1], a2 = b[2];
        for (int c = 0; c < Cin; ++c) {
            float v = src[(long)c * HW];
            a0 += w[c] * v;
            a1 += w[Cin + c] * v;
            a2 += w[2 * Cin + c] * v;
        }
        out[(bb * 3 + 0) * HW + p] = a0;
        out[(bb * 3 + 1) * HW + p] = a1;
        out[(bb * 3 + 2) * HW + p] = a2;
    }
}

// ---------------------------------------------------------------------------
// out = mean_c(x_out) * x / (mean_c(x) + 1e-6)
// ---------------------------------------------------------------------------
__global__ void k_final(const float* __restrict__ x, const float* __restrict__ xo,
                        float* __restrict__ out, int B, int HW) {
    const long n = (long)B * HW;
    for (long i = blockIdx.x * (long)blockDim.x + threadIdx.x; i < n;
         i += (long)gridDim.x * blockDim.x) {
        long b = i / HW, p = i % HW;
        float x0 = x[(b * 3 + 0) * HW + p];
        float x1 = x[(b * 3 + 1) * HW + p];
        float x2 = x[(b * 3 + 2) * HW + p];
        float o0 = xo[(b * 3 + 0) * HW + p];
        float o1 = xo[(b * 3 + 1) * HW + p];
        float o2 = xo[(b * 3 + 2) * HW + p];
        float xm = (x0 + x1 + x2) * (1.f / 3.f);
        float om = (o0 + o1 + o2) * (1.f / 3.f);
        float s  = om / (xm + 1e-6f);
        out[(b * 3 + 0) * HW + p] = x0 * s;
        out[(b * 3 + 1) * HW + p] = x1 * s;
        out[(b * 3 + 2) * HW + p] = x2 * s;
    }
}

// ---------------------------------------------------------------------------
// Orchestration
// ---------------------------------------------------------------------------
extern "C" void kernel_launch(void* const* d_in, const int* in_sizes, int n_in,
                              void* d_out, int out_size, void* d_ws, size_t ws_size,
                              hipStream_t stream) {
    (void)in_sizes; (void)n_in; (void)out_size; (void)ws_size;
    const float* x      = (const float*)d_in[0];
    const float* n16    = (const float*)d_in[1];
    const float* sig16  = (const float*)d_in[2];
    const float* bia16  = (const float*)d_in[3];
    const float* prelua = (const float*)d_in[4];
    const float* bn3g   = (const float*)d_in[5];
    const float* bn3b   = (const float*)d_in[6];
    const float* bn32g  = (const float*)d_in[7];
    const float* bn32b  = (const float*)d_in[8];
    const float* bn64g  = (const float*)d_in[9];
    const float* bn64b  = (const float*)d_in[10];
    const float* bn128g = (const float*)d_in[11];
    const float* bn128b = (const float*)d_in[12];
    const float* w1 = (const float*)d_in[13]; const float* b1 = (const float*)d_in[14];
    const float* w2 = (const float*)d_in[15]; const float* b2 = (const float*)d_in[16];
    const float* w3 = (const float*)d_in[17]; const float* b3 = (const float*)d_in[18];
    const float* w4 = (const float*)d_in[19]; const float* b4 = (const float*)d_in[20];
    const float* w5 = (const float*)d_in[21]; const float* b5 = (const float*)d_in[22];
    const float* w6 = (const float*)d_in[23]; const float* b6 = (const float*)d_in[24];
    const float* w7 = (const float*)d_in[25]; const float* b7 = (const float*)d_in[26];
    const float* w11 = (const float*)d_in[27]; const float* b11 = (const float*)d_in[28];

    const int B = 8, H = 256, W = 256, HW = H * W;

    char* ws = (char*)d_ws;
    size_t off = 0;
    auto carve = [&](size_t bytes) -> char* {
        char* p = ws + off;
        off = (off + bytes + 255) & ~(size_t)255;
        return p;
    };
    f16*   stage = (f16*)  carve((size_t)B * 64 * HW * 2);        // 67 MB, reused per conv
    float* convo = (float*)carve((size_t)B * 32 * HW * 4);        // 67 MB scratch conv out
    float* abig  = (float*)carve((size_t)B * 32 * HW * 4);        // 67 MB sigmoid activations
    float* x1b   = (float*)carve((size_t)B * 32 * (HW / 4) * 4);
    float* x2b   = (float*)carve((size_t)B * 64 * (HW / 16) * 4);
    float* x3b   = (float*)carve((size_t)B * 128 * (HW / 64) * 4);
    float* x4b   = (float*)carve((size_t)B * 128 * (HW / 64) * 4);
    float* x5b   = (float*)carve((size_t)B * 64 * (HW / 16) * 4);
    float* x6b   = (float*)carve((size_t)B * 32 * (HW / 4) * 4);
    float* xob   = (float*)carve((size_t)B * 3 * HW * 4);
    f16* w1p = (f16*)carve(9L * 32 * 64 * 2);
    f16* w2p = (f16*)carve(9L * 64 * 32 * 2);
    f16* w3p = (f16*)carve(9L * 128 * 64 * 2);
    f16* w4p = (f16*)carve(9L * 128 * 128 * 2);
    f16* w5p = (f16*)carve(9L * 64 * 256 * 2);
    f16* w6p = (f16*)carve(9L * 32 * 128 * 2);
    f16* w7p = (f16*)carve(9L * 32 * 64 * 2);
    f16* zbuf = (f16*)carve(1024 * 2);       // zero page for OOB conv taps
    float* stats = (float*)carve(256 * 4);

    auto ew = [](long n) { return dim3((unsigned)((n + 255) / 256)); };

    // ---- weight prep (f32 OIHW -> f16 [tap][O][Ipad]) + zero page ----
    k_wprep<<<ew(9L * 32 * 64), 256, 0, stream>>>(w1, w1p, 32, 48, 64);
    k_wprep<<<ew(9L * 64 * 32), 256, 0, stream>>>(w2, w2p, 64, 32, 32);
    k_wprep<<<ew(9L * 128 * 64), 256, 0, stream>>>(w3, w3p, 128, 64, 64);
    k_wprep<<<ew(9L * 128 * 128), 256, 0, stream>>>(w4, w4p, 128, 128, 128);
    k_wprep<<<ew(9L * 64 * 256), 256, 0, stream>>>(w5, w5p, 64, 256, 256);
    k_wprep<<<ew(9L * 32 * 128), 256, 0, stream>>>(w6, w6p, 32, 128, 128);
    k_wprep<<<ew(9L * 32 * 64), 256, 0, stream>>>(w7, w7p, 32, 64, 64);
    k_zero_f16<<<1, 256, 0, stream>>>(zbuf, 1024);

    // ---- curve stage: stats -> scale/shift -> fused f16 NHWC x_fusion ----
    k_zero_f32<<<1, 256, 0, stream>>>(stats, 96);
    k_curve_stats<<<dim3(512, 3), 256, 0, stream>>>(x, n16, sig16, bia16, stats, B, HW);
    k_bn_finalize<<<1, 48, 0, stream>>>(stats, bn3g, bn3b, 48, 3, 1.f / ((float)B * HW));
    k_zero_f16<<<ew((long)B * 64 * HW), 256, 0, stream>>>(stage, (long)B * 64 * HW);
    k_curve_apply<<<dim3(512, 3), 256, 0, stream>>>(x, n16, sig16, bia16, stats, stage, B, HW);

    // ---- conv1 (64pad -> 32, 256x256) + prelu/pool + BN -> x1 ----
    k_conv3x3_wmma<<<dim3(4096, 1), 256, 0, stream>>>(stage, w1p, b1, zbuf, convo, B, 64, 32, 256, 256);
    k_prelu_pool<<<ew((long)B * 32 * 128 * 128), 256, 0, stream>>>(convo, x1b, prelua, B, 32, 256, 256);
    k_zero_f32<<<1, 256, 0, stream>>>(stats, 64);
    k_chan_stats<<<dim3(256, 32), 256, 0, stream>>>(x1b, stats, B, 32, 128 * 128);
    k_bn_finalize<<<1, 32, 0, stream>>>(stats, bn32g, bn32b, 32, 32, 1.f / (8.f * 128 * 128));
    k_bn_apply<<<ew((long)B * 32 * 16384), 256, 0, stream>>>(x1b, x1b, stats, (long)B * 32 * 16384, 32, 16384);

    // ---- conv2 (32 -> 64, 128x128) + prelu/pool + BN -> x2 ----
    k_cast_slice_f16<<<ew((long)B * 32 * 16384), 256, 0, stream>>>(x1b, stage, B, 32, 32, 0, 16384);
    k_conv3x3_wmma<<<dim3(1024, 2), 256, 0, stream>>>(stage, w2p, b2, zbuf, convo, B, 32, 64, 128, 128);
    k_prelu_pool<<<ew((long)B * 64 * 64 * 64), 256, 0, stream>>>(convo, x2b, prelua, B, 64, 128, 128);
    k_zero_f32<<<1, 256, 0, stream>>>(stats, 128);
    k_chan_stats<<<dim3(256, 64), 256, 0, stream>>>(x2b, stats, B, 64, 64 * 64);
    k_bn_finalize<<<1, 64, 0, stream>>>(stats, bn64g, bn64b, 64, 64, 1.f / (8.f * 64 * 64));
    k_bn_apply<<<ew((long)B * 64 * 4096), 256, 0, stream>>>(x2b, x2b, stats, (long)B * 64 * 4096, 64, 4096);

    // ---- conv3 (64 -> 128, 64x64) + prelu/pool + BN -> x3 ----
    k_cast_slice_f16<<<ew((long)B * 64 * 4096), 256, 0, stream>>>(x2b, stage, B, 64, 64, 0, 4096);
    k_conv3x3_wmma<<<dim3(256, 4), 256, 0, stream>>>(stage, w3p, b3, zbuf, convo, B, 64, 128, 64, 64);
    k_prelu_pool<<<ew((long)B * 128 * 32 * 32), 256, 0, stream>>>(convo, x3b, prelua, B, 128, 64, 64);
    k_zero_f32<<<1, 256, 0, stream>>>(stats, 256);
    k_chan_stats<<<dim3(64, 128), 256, 0, stream>>>(x3b, stats, B, 128, 32 * 32);
    k_bn_finalize<<<1, 128, 0, stream>>>(stats, bn128g, bn128b, 128, 128, 1.f / (8.f * 32 * 32));
    k_bn_apply<<<ew((long)B * 128 * 1024), 256, 0, stream>>>(x3b, x3b, stats, (long)B * 128 * 1024, 128, 1024);

    // ---- conv4 (128 -> 128, 32x32) + BN -> x4 (no prelu/pool) ----
    k_cast_slice_f16<<<ew((long)B * 128 * 1024), 256, 0, stream>>>(x3b, stage, B, 128, 128, 0, 1024);
    k_conv3x3_wmma<<<dim3(64, 4), 256, 0, stream>>>(stage, w4p, b4, zbuf, convo, B, 128, 128, 32, 32);
    k_zero_f32<<<1, 256, 0, stream>>>(stats, 256);
    k_chan_stats<<<dim3(64, 128), 256, 0, stream>>>(convo, stats, B, 128, 1024);
    k_bn_finalize<<<1, 128, 0, stream>>>(stats, bn128g, bn128b, 128, 128, 1.f / (8.f * 1024));
    k_bn_apply<<<ew((long)B * 128 * 1024), 256, 0, stream>>>(convo, x4b, stats, (long)B * 128 * 1024, 128, 1024);

    // ---- conv5 (cat[x4,x3]=256 -> 64, 32x32) + prelu + up2 + BN -> x5 ----
    k_cast_slice_f16<<<ew((long)B * 128 * 1024), 256, 0, stream>>>(x4b, stage, B, 128, 256, 0, 1024);
    k_cast_slice_f16<<<ew((long)B * 128 * 1024), 256, 0, stream>>>(x3b, stage, B, 128, 256, 128, 1024);
    k_conv3x3_wmma<<<dim3(64, 2), 256, 0, stream>>>(stage, w5p, b5, zbuf, convo, B, 256, 64, 32, 32);
    k_up2_prelu<<<ew((long)B * 64 * 4096), 256, 0, stream>>>(convo, x5b, prelua, B, 64, 32, 32, 0);
    k_zero_f32<<<1, 256, 0, stream>>>(stats, 128);
    k_chan_stats<<<dim3(256, 64), 256, 0, stream>>>(x5b, stats, B, 64, 4096);
    k_bn_finalize<<<1, 64, 0, stream>>>(stats, bn64g, bn64b, 64, 64, 1.f / (8.f * 4096));
    k_bn_apply<<<ew((long)B * 64 * 4096), 256, 0, stream>>>(x5b, x5b, stats, (long)B * 64 * 4096, 64, 4096);

    // ---- conv6 (cat[x2,x5]=128 -> 32, 64x64) + prelu + up2 + BN -> x6 ----
    k_cast_slice_f16<<<ew((long)B * 64 * 4096), 256, 0, stream>>>(x2b, stage, B, 64, 128, 0, 4096);
    k_cast_slice_f16<<<ew((long)B * 64 * 4096), 256, 0, stream>>>(x5b, stage, B, 64, 128, 64, 4096);
    k_conv3x3_wmma<<<dim3(256, 1), 256, 0, stream>>>(stage, w6p, b6, zbuf, convo, B, 128, 32, 64, 64);
    k_up2_prelu<<<ew((long)B * 32 * 16384), 256, 0, stream>>>(convo, x6b, prelua, B, 32, 64, 64, 0);
    k_zero_f32<<<1, 256, 0, stream>>>(stats, 64);
    k_chan_stats<<<dim3(256, 32), 256, 0, stream>>>(x6b, stats, B, 32, 16384);
    k_bn_finalize<<<1, 32, 0, stream>>>(stats, bn32g, bn32b, 32, 32, 1.f / (8.f * 16384));
    k_bn_apply<<<ew((long)B * 32 * 16384), 256, 0, stream>>>(x6b, x6b, stats, (long)B * 32 * 16384, 32, 16384);

    // ---- conv7 (cat[x1,x6]=64 -> 32, 128x128) + prelu + up2 + sigmoid ----
    k_cast_slice_f16<<<ew((long)B * 32 * 16384), 256, 0, stream>>>(x1b, stage, B, 32, 64, 0, 16384);
    k_cast_slice_f16<<<ew((long)B * 32 * 16384), 256, 0, stream>>>(x6b, stage, B, 32, 64, 32, 16384);
    k_conv3x3_wmma<<<dim3(1024, 1), 256, 0, stream>>>(stage, w7p, b7, zbuf, convo, B, 64, 32, 128, 128);
    k_up2_prelu<<<ew((long)B * 32 * HW), 256, 0, stream>>>(convo, abig, prelua, B, 32, 128, 128, 1);

    // ---- 1x1 conv (32 -> 3) and final recombination ----
    k_conv1x1_3<<<ew((long)B * HW), 256, 0, stream>>>(abig, w11, b11, xob, B, 32, HW);
    k_final<<<ew((long)B * HW), 256, 0, stream>>>(x, xob, (float*)d_out, B, HW);
}
